// FourierScaledDotProductAttention_71305047048411
// MI455X (gfx1250) — compile-verified, matrix operand order checked by reading, not run on hardware
//
#include <hip/hip_runtime.h>

typedef _Float16 f16;
typedef __attribute__((ext_vector_type(16))) _Float16 v16h;
typedef __attribute__((ext_vector_type(8)))  float    v8f;
typedef __attribute__((ext_vector_type(4)))  unsigned v4u;
typedef __attribute__((ext_vector_type(8)))  int      v8i;
typedef __attribute__((ext_vector_type(4)))  int      v4i;

#define D      512
#define BQ     64     // query rows per block
#define BC     32     // kv rows per inner tile
#define S_LEN  2048
#define B_BATCH 8

// K index for VGPR v, lane-half h in the 16-bit A/B WMMA layouts (ISA 7.12.2)
__device__ __forceinline__ int kof(int v, int h) {
    return (v < 4) ? (2*v + 8*h) : (16 + 2*(v-4) + 8*h);
}

// ---- Tensor Data Mover: issue one global->LDS tile DMA (ISA ch.8 D# layout) ----
// 2D tile: tile0 elements (8B units) per row, tile1 rows, row stride stride0 (8B units).
__device__ __forceinline__ void tdm_load(const void* gptr, void* lptr,
                                         unsigned tile0, unsigned tile1,
                                         unsigned long long stride0) {
    unsigned long long ga = (unsigned long long)(uintptr_t)gptr;
    unsigned long long la = (unsigned long long)(uintptr_t)lptr;   // low 32 = LDS offset
    v4u g0;
    g0[0] = 1u;                                            // count=1, user descriptor
    g0[1] = (unsigned)la;                                  // lds_addr
    g0[2] = (unsigned)ga;                                  // global_addr[31:0]
    g0[3] = (unsigned)((ga >> 32) & 0x01FFFFFFu) | (2u << 30);  // addr[56:32], type=2
    unsigned long long td0 = 0x40000000ull, td1 = 0x40000000ull; // huge -> no OOB clip
    unsigned long long u0 = (3ull << 16)                   // data_size = 8B
                          | ((td0 & 0xFFFFull) << 48);     // tensor_dim0[15:0]
    unsigned long long u1 = ((td0 >> 16) & 0xFFFFull)      // tensor_dim0[31:16]
                          | ((td1 & 0xFFFFFFFFull) << 16)  // tensor_dim1
                          | ((unsigned long long)(tile0 & 0xFFFFu) << 48);
    unsigned long long u2 = (unsigned long long)(tile1 & 0xFFFFu)   // tile_dim1
                          | ((stride0 & 0xFFFFFFFFull) << 32);      // dim0_stride[31:0]
    unsigned long long u3 = (stride0 >> 32) & 0xFFFFull;            // dim0_stride[47:32]
    v8i g1;
    g1[0]=(int)u0; g1[1]=(int)(u0>>32); g1[2]=(int)u1; g1[3]=(int)(u1>>32);
    g1[4]=(int)u2; g1[5]=(int)(u2>>32); g1[6]=(int)u3; g1[7]=(int)(u3>>32);
    v4i gz = {0,0,0,0};
#if defined(__clang_major__) && (__clang_major__ >= 23)
    v8i gz8 = {0,0,0,0,0,0,0,0};
    __builtin_amdgcn_tensor_load_to_lds(g0, g1, gz, gz, gz8, 0);
#else
    __builtin_amdgcn_tensor_load_to_lds(g0, g1, gz, gz, 0);
#endif
}

// ---------------- kernel 0: circulant vectors c_q,c_k,c_v ----------------
__global__ void k_circ(const float* qr, const float* qi, const float* kr,
                       const float* ki, const float* vr, const float* vi,
                       float* cbuf) {
    int t = blockIdx.x * blockDim.x + threadIdx.x;
    if (t >= 3 * D) return;
    int f = t / D, d = t % D;
    const float* re = (f == 0) ? qr : (f == 1) ? kr : vr;
    const float* im = (f == 0) ? qi : (f == 1) ? ki : vi;
    float s = 0.f;
    for (int k = 0; k < D; ++k) {
        int r = (k * d) & (D - 1);                       // exact k*d mod 512
        float ang = 6.283185307179586f * (float)r * (1.0f / (float)D);
        s += re[k] * __cosf(ang) - im[k] * __sinf(ang);
    }
    cbuf[t] = s * (1.0f / (float)D);
}

// -------- kernel 1: expand c into WMMA-B-swizzled f16 filter matrices -----
__global__ void k_fill(const float* cbuf, f16* filt) {
    int flat = blockIdx.x * blockDim.x + threadIdx.x;
    if (flat >= 3 * D * D) return;
    int e    =  flat        & 15;
    int lane = (flat >>  4) & 31;
    int kt   = (flat >>  9) & 15;
    int nt   = (flat >> 13) & 31;
    int f    =  flat >> 18;
    int half = lane >> 4, nl = lane & 15;
    int kloc = kof(e >> 1, half) + (e & 1);
    int k = kt * 32 + kloc, n = nt * 16 + nl;
    filt[flat] = (f16)cbuf[f * D + ((n - k) & (D - 1))];
}

// ---------------- kernel 2: q/k/v = x @ M^T  (f16 WMMA GEMM) --------------
// q,k row-major [b*S+s][d];  v stored TRANSPOSED: vw[(b*D+d)*S + s]
__global__ __launch_bounds__(256) void k_qkv(const float* __restrict__ x,
                                             const f16* __restrict__ filt,
                                             f16* qw, f16* kw, f16* vw) {
    __shared__ f16 Xs[BQ][D];                            // 64 KB
    int tid = threadIdx.x, w = tid >> 5, lane = tid & 31;
    int nloc = lane & 15, half = lane >> 4;
    size_t rowbase = (size_t)blockIdx.x * BQ;

    const float4* xsrc = (const float4*)(x + rowbase * D);
    f16* xf = &Xs[0][0];
    for (int i = tid; i < BQ * D / 4; i += 256) {
        float4 t = xsrc[i];
        xf[4*i+0] = (f16)t.x; xf[4*i+1] = (f16)t.y;
        xf[4*i+2] = (f16)t.z; xf[4*i+3] = (f16)t.w;
    }
    __syncthreads();

    for (int f = 0; f < 3; ++f) {
        const f16* fb = filt + (size_t)f * D * D;
        for (int mi = 0; mi < 4; ++mi) {
            for (int ntl = 0; ntl < 4; ++ntl) {
                int nt = w * 4 + ntl;
                v8f acc = {};
                for (int kt = 0; kt < 16; ++kt) {
                    union { unsigned u[8]; v16h v; } A;
                    int m = mi * 16 + nloc;
#pragma unroll
                    for (int vv = 0; vv < 8; ++vv)
                        A.u[vv] = *(const unsigned*)&Xs[m][kt*32 + kof(vv, half)];
                    v16h Bv = *(const v16h*)(fb + (((size_t)nt*16 + kt)*32 + lane)*16);
                    acc = __builtin_amdgcn_wmma_f32_16x16x32_f16(
                        false, A.v, false, Bv, (short)0, acc, false, false);
                }
#pragma unroll
                for (int r = 0; r < 8; ++r) {
                    size_t row = rowbase + mi*16 + r + half*8;   // global flat s-row
                    int col = nt*16 + nloc;
                    if (f == 0) {
                        qw[row * D + col] = (f16)acc[r];
                    } else if (f == 1) {
                        kw[row * D + col] = (f16)acc[r];
                    } else {                    // V transposed: [b][d][s]
                        size_t bb = row / S_LEN, ss = row % S_LEN;
                        vw[(bb * D + col) * S_LEN + ss] = (f16)acc[r];
                    }
                }
            }
        }
    }
}

// -------- kernel 3: flash attention, Br=64/Bc=32, TDM double-buffered -----
__global__ __launch_bounds__(256) void k_attn(const f16* __restrict__ qw,
                                              const f16* __restrict__ kw,
                                              const f16* __restrict__ vw,
                                              float* __restrict__ out) {
    __shared__ __align__(64) f16  Ks[2][BC][D];   // 2 x 32 KB
    __shared__ __align__(64) f16  Vt[2][D][BC];   // 2 x 32 KB (transposed)
    __shared__ float Ss[BQ][BC];                  //  8 KB
    __shared__ f16   Ps[BQ][BC];                  //  4 KB
    __shared__ float sAlpha[BQ], sM[BQ], sL[BQ];

    int tid = threadIdx.x, w = tid >> 5, lane = tid & 31;
    int nloc = lane & 15, half = lane >> 4;
    int b  = blockIdx.x / (S_LEN / BQ);
    int qb = (blockIdx.x % (S_LEN / BQ)) * BQ;
    int smi = w & 3, sni = w >> 2;    // score-tile ownership
    const int NJ = S_LEN / BC;

    // ---- hoist this wave's Q A-fragments (16 x v16h = 128 VGPRs) from global ----
    v16h Qf[16];
    {
        const f16* qrow = qw + ((size_t)b*S_LEN + qb + smi*16 + nloc) * D;
#pragma unroll
        for (int kt = 0; kt < 16; ++kt) {
            union { unsigned u[8]; v16h v; } A;
#pragma unroll
            for (int vv = 0; vv < 8; ++vv)
                A.u[vv] = *(const unsigned*)(qrow + kt*32 + kof(vv, half));
            Qf[kt] = A.v;
        }
    }

    if (tid < BQ) { sM[tid] = -1e30f; sL[tid] = 0.f; }

    v8f O[4][4];
    v8f vzero = {};
#pragma unroll
    for (int mi = 0; mi < 4; ++mi)
#pragma unroll
        for (int dt = 0; dt < 4; ++dt) O[mi][dt] = vzero;

    // ---- TDM prologue: stage tile 0 ----
    if (w == 0) {
        tdm_load(kw + ((size_t)b*S_LEN) * D, &Ks[0][0][0], 4096, 0, 4096ull);
        tdm_load(vw + ((size_t)b*D) * S_LEN, &Vt[0][0][0], 8, 512, 512ull);
    }

    const float scale = 0.044194173824159216f;  // 1/sqrt(512)

    for (int j = 0; j < NJ; ++j) {
        int cur = j & 1;
        if (w == 0) {
            if (j + 1 < NJ) {   // async: stage tile j+1 into spare buffer
                tdm_load(kw + ((size_t)b*S_LEN + (size_t)(j+1)*BC) * D,
                         &Ks[cur ^ 1][0][0], 4096, 0, 4096ull);
                tdm_load(vw + ((size_t)b*D) * S_LEN + (size_t)(j+1)*BC,
                         &Vt[cur ^ 1][0][0], 8, 512, 512ull);
                __builtin_amdgcn_s_wait_tensorcnt(2);   // tile j complete (in-order)
            } else {
                __builtin_amdgcn_s_wait_tensorcnt(0);
            }
        }
        __syncthreads();

        // ---- scores: each wave one 16x16 tile of S = Q K^T ----
        v8f sc = {};
        for (int kt = 0; kt < 16; ++kt) {
            union { unsigned u[8]; v16h v; } Bf;
#pragma unroll
            for (int vv = 0; vv < 8; ++vv)
                Bf.u[vv] = *(const unsigned*)&Ks[cur][sni*16 + nloc][kt*32 + kof(vv, half)];
            sc = __builtin_amdgcn_wmma_f32_16x16x32_f16(
                false, Qf[kt], false, Bf.v, (short)0, sc, false, false);
        }
#pragma unroll
        for (int r = 0; r < 8; ++r)
            Ss[smi*16 + r + half*8][sni*16 + nloc] = sc[r];
        __syncthreads();

        // ---- online softmax: one thread per query row ----
        if (tid < BQ) {
            float m_old = sM[tid], bm = -1e30f;
            float vals[BC];
#pragma unroll
            for (int c = 0; c < BC; ++c) { vals[c] = Ss[tid][c] * scale; bm = fmaxf(bm, vals[c]); }
            float mn = fmaxf(m_old, bm);
            float alpha = __expf(m_old - mn);
            float sum = 0.f;
#pragma unroll
            for (int c = 0; c < BC; ++c) {
                float p = __expf(vals[c] - mn);
                sum += p; Ps[tid][c] = (f16)p;
            }
            sM[tid] = mn; sL[tid] = sL[tid] * alpha + sum; sAlpha[tid] = alpha;
        }
        __syncthreads();

        // ---- rescale accumulators + O += P @ V ----
#pragma unroll
        for (int mi = 0; mi < 4; ++mi) {
            float al[8];
#pragma unroll
            for (int r = 0; r < 8; ++r) al[r] = sAlpha[mi*16 + r + half*8];
            union { unsigned u[8]; v16h v; } A;
#pragma unroll
            for (int vv = 0; vv < 8; ++vv)
                A.u[vv] = *(const unsigned*)&Ps[mi*16 + nloc][kof(vv, half)];
#pragma unroll
            for (int dt4 = 0; dt4 < 4; ++dt4) {
                int dt = w * 4 + dt4;
#pragma unroll
                for (int r = 0; r < 8; ++r) O[mi][dt4][r] *= al[r];
                union { unsigned u[8]; v16h v; } Bf;     // V^T: pair-contiguous in LDS
#pragma unroll
                for (int vv = 0; vv < 8; ++vv)
                    Bf.u[vv] = *(const unsigned*)&Vt[cur][dt*16 + nloc][kof(vv, half)];
                O[mi][dt4] = __builtin_amdgcn_wmma_f32_16x16x32_f16(
                    false, A.v, false, Bf.v, (short)0, O[mi][dt4], false, false);
            }
        }
        __syncthreads();   // frees buf[cur] for the TDM issued next iteration
    }

    // ---- normalize and write ----
#pragma unroll
    for (int mi = 0; mi < 4; ++mi) {
        float li[8];
#pragma unroll
        for (int r = 0; r < 8; ++r) li[r] = 1.f / sL[mi*16 + r + half*8];
#pragma unroll
        for (int dt4 = 0; dt4 < 4; ++dt4) {
            int dt = w * 4 + dt4;
#pragma unroll
            for (int r = 0; r < 8; ++r) {
                size_t row = (size_t)b*S_LEN + qb + mi*16 + r + half*8;
                out[row * D + dt*16 + nloc] = O[mi][dt4][r] * li[r];
            }
        }
    }
}

extern "C" void kernel_launch(void* const* d_in, const int* in_sizes, int n_in,
                              void* d_out, int out_size, void* d_ws, size_t ws_size,
                              hipStream_t stream) {
    const float* x  = (const float*)d_in[0];
    const float* qr = (const float*)d_in[1];
    const float* qi = (const float*)d_in[2];
    const float* kr = (const float*)d_in[3];
    const float* ki = (const float*)d_in[4];
    const float* vr = (const float*)d_in[5];
    const float* vi = (const float*)d_in[6];

    char* ws = (char*)d_ws;
    float* cbuf = (float*)ws;                             // 3*512 f32
    f16* filt = (f16*)(ws + 8192);                        // 1.5 MB swizzled B mats
    f16* qw = (f16*)(ws + (size_t)(2u << 20));            // 16 MB each
    f16* kw = (f16*)(ws + (size_t)(2u << 20) + (size_t)(16u << 20));
    f16* vw = (f16*)(ws + (size_t)(2u << 20) + (size_t)(32u << 20));

    k_circ<<<(3 * D + 255) / 256, 256, 0, stream>>>(qr, qi, kr, ki, vr, vi, cbuf);
    k_fill<<<(3 * D * D + 255) / 256, 256, 0, stream>>>(cbuf, filt);
    k_qkv<<<(B_BATCH * S_LEN) / BQ, 256, 0, stream>>>(x, filt, qw, kw, vw);
    k_attn<<<B_BATCH * (S_LEN / BQ), 256, 0, stream>>>(qw, kw, vw, (float*)d_out);
}
